// GraphTransformer_67929202753592
// MI455X (gfx1250) — matrix operand order repaired and checked by reading, not compiled, not run on hardware
//
#include <hip/hip_runtime.h>
#include <hip/hip_bf16.h>
#include <math.h>

#define NN 50000
#define NE 800000
#define NG 512
#define HCC 256
#define QC 1024

typedef __attribute__((ext_vector_type(16))) _Float16 v16h;
typedef __attribute__((ext_vector_type(8)))  _Float16 v8h;
typedef __attribute__((ext_vector_type(8)))  float    v8f;

// single shared dynamic-LDS symbol (no static LDS in kernels that use it,
// so it sits at LDS byte offset 0 -- required by the async-copy asm below)
extern __shared__ __align__(16) unsigned char dynsmem[];

__device__ __forceinline__ void atomic_max_f32(float* addr, float v) {
  // monotone bit-pattern trick; m initialized to -inf (0xFF800000)
  if (v >= 0.0f) atomicMax((int*)addr, __float_as_int(v));
  else           atomicMin((unsigned int*)addr, __float_as_uint(v));
}

// async 16-byte global->LDS copy (CDNA5, tracked by ASYNCcnt)
__device__ __forceinline__ void async_copy16(unsigned int lds_byte_off, const void* gptr) {
  unsigned long long ga = (unsigned long long)(uintptr_t)gptr;
  asm volatile("global_load_async_to_lds_b128 %0, %1, off"
               :: "v"(lds_byte_off), "v"(ga) : "memory");
}
__device__ __forceinline__ void async_wait0() {
  asm volatile("s_wait_asynccnt 0x0" ::: "memory");
}

// ---------------- weight packing: WcatT[n][k] f16 (col-major), bias[n] ----
__global__ void pack_weights_kernel(const float* __restrict__ Wq, const float* __restrict__ Wk,
                                    const float* __restrict__ Wv, const float* __restrict__ Ws,
                                    const float* __restrict__ bq, const float* __restrict__ bk,
                                    const float* __restrict__ bv, const float* __restrict__ bs,
                                    _Float16* __restrict__ wpackT, float* __restrict__ bpack, int K) {
  int idx = blockIdx.x * blockDim.x + threadIdx.x;
  int total = QC * K;
  if (idx >= total) return;
  int n = idx / K, k = idx - n * K;
  int sel = n >> 8, col = n & 255;
  const float* W = (sel == 0) ? Wq : (sel == 1) ? Wk : (sel == 2) ? Wv : Ws;
  wpackT[(size_t)n * K + k] = (_Float16)W[k * HCC + col];
  if (k == 0) {
    const float* b = (sel == 0) ? bq : (sel == 1) ? bk : (sel == 2) ? bv : bs;
    bpack[n] = b[col];
  }
}

// ---------------- WMMA GEMM: out[M,1024] = X[M,K] @ Wcat + bias ------------
// block = 256 thr = 8 waves; block tile 16(M) x 1024(N); wave tile 16 x 128.
// A tile staged in LDS as f16 once per block (<= 8KB), fragments via ds_load.
__global__ __launch_bounds__(256)
void gemm_qkvs_kernel(const float* __restrict__ X, const _Float16* __restrict__ WT,
                      const float* __restrict__ bias, float* __restrict__ out,
                      int nrows, int K, int kshift) {
  _Float16* sA = (_Float16*)dynsmem;   // [16][K] f16
  const int m0 = blockIdx.x * 16;

  // cooperative fp32->f16 staging of the A tile
  for (int idx = threadIdx.x; idx < (16 << kshift); idx += 256) {
    int r = idx >> kshift, k = idx - (r << kshift);
    int row = m0 + r; if (row >= nrows) row = nrows - 1;
    sA[idx] = (_Float16)X[(size_t)row * K + k];
  }
  __syncthreads();

  const int lane = threadIdx.x & 31;
  const int wave = threadIdx.x >> 5;   // 0..7
  const int g    = lane >> 4;          // half-wave group
  const int lm   = lane & 15;
  const int nbase = wave * 128;

  v8f acc[8] = {};

  for (int kk = 0; kk < K; kk += 32) {
    // ---- A fragment (contiguous k order, k = kk + 16*g + s) from LDS ----
    const _Float16* ap = sA + (lm << kshift) + kk + g * 16;
    v8h lo = *(const v8h*)(ap);        // k = kk + g*16 .. +7
    v8h hi = *(const v8h*)(ap + 8);    // k = kk + g*16 + 8 .. +15
    v16h a = __builtin_shufflevector(lo, hi, 0,1,2,3,4,5,6,7,8,9,10,11,12,13,14,15);

    #pragma unroll
    for (int t = 0; t < 8; ++t) {
      int n = nbase + t * 16 + lm;
      // B fragment: 16 consecutive k at fixed column (col-major pack), 32B aligned
      v16h b = *(const v16h*)(WT + (size_t)n * K + kk + g * 16);
      acc[t] = __builtin_amdgcn_wmma_f32_16x16x32_f16(false, a, false, b,
                                                      (short)0, acc[t], false, false);
    }
  }
  // ---- epilogue: C layout: VGPR r -> M = r + 8*g, N = lane&15 ----
  #pragma unroll
  for (int t = 0; t < 8; ++t) {
    int n = nbase + t * 16 + lm;
    float bv = bias[n];
    #pragma unroll
    for (int r = 0; r < 8; ++r) {
      int row = m0 + r + 8 * g;
      if (row < nrows) out[(size_t)row * QC + n] = acc[t][r] + bv;
    }
  }
}

// ---------------- per-layer accumulator init --------------------------------
__global__ void init_layer_kernel(float* __restrict__ mbuf, float* __restrict__ sbuf,
                                  float* __restrict__ accb) {
  int idx = blockIdx.x * blockDim.x + threadIdx.x;
  if (idx < NN * 4) { mbuf[idx] = -__builtin_inff(); sbuf[idx] = 0.0f; }
  if (idx < NN * HCC) accb[idx] = 0.0f;
}

// stage We[16][256] + be[256] into LDS via CDNA5 async global->LDS copies
__device__ __forceinline__ void stage_We_async(const float* __restrict__ We,
                                               const float* __restrict__ be) {
  int t = threadIdx.x;
  #pragma unroll
  for (int i = 0; i < 4; ++i) {
    int fidx = (i * 256 + t) * 4;                       // float index into We
    async_copy16((unsigned int)(fidx * 4), We + fidx);  // LDS bytes [0, 16384)
  }
  if (t < 64) {
    int fidx = t * 4;
    async_copy16((unsigned int)(16 * HCC * 4 + fidx * 4), be + fidx); // [16384, 17408)
  }
  async_wait0();       // wave's own async copies done
  __syncthreads();     // all waves' copies visible
}

// per-lane edge embedding for 8 channels starting at c0 (reads LDS copy)
__device__ __forceinline__ void edge_embed8(const float* __restrict__ sWe,
                                            const float* __restrict__ sbe,
                                            const float* ed, int c0, float* ee) {
  #pragma unroll
  for (int j = 0; j < 8; ++j) ee[j] = sbe[c0 + j];
  #pragma unroll
  for (int d = 0; d < 16; ++d) {
    float w = ed[d];
    float4 w0 = *(const float4*)(sWe + d * HCC + c0);
    float4 w1 = *(const float4*)(sWe + d * HCC + c0 + 4);
    ee[0] += w * w0.x; ee[1] += w * w0.y; ee[2] += w * w0.z; ee[3] += w * w0.w;
    ee[4] += w * w1.x; ee[5] += w * w1.y; ee[6] += w * w1.z; ee[7] += w * w1.w;
  }
}

// ---------------- edge pass A: logits + segment max -------------------------
__global__ __launch_bounds__(256)
void edge_logits_kernel(const int* __restrict__ eidx, const float* __restrict__ ea,
                        const float* __restrict__ We, const float* __restrict__ be,
                        const float* __restrict__ qkvs,
                        float* __restrict__ logits, float* __restrict__ mbuf) {
  const float* sWe = (const float*)dynsmem;
  const float* sbe = sWe + 16 * HCC;
  stage_We_async(We, be);

  int lane = threadIdx.x & 31;
  int e = blockIdx.x * 8 + (threadIdx.x >> 5);
  if (e >= NE) return;
  int src = eidx[e];
  int dst = eidx[NE + e];

  float my = (lane < 16) ? ea[(size_t)e * 16 + lane] : 0.0f;
  float ed[16];
  #pragma unroll
  for (int d = 0; d < 16; ++d) ed[d] = __shfl(my, d, 32);

  int c0 = lane * 8;
  float ee[8];
  edge_embed8(sWe, sbe, ed, c0, ee);

  const float* qrow = qkvs + (size_t)dst * QC + c0;         // Q block
  const float* krow = qkvs + (size_t)src * QC + 256 + c0;   // K block
  float4 q0 = *(const float4*)qrow,        q1 = *(const float4*)(qrow + 4);
  float4 k0 = *(const float4*)krow,        k1 = *(const float4*)(krow + 4);

  float part = q0.x*(k0.x+ee[0]) + q0.y*(k0.y+ee[1]) + q0.z*(k0.z+ee[2]) + q0.w*(k0.w+ee[3])
             + q1.x*(k1.x+ee[4]) + q1.y*(k1.y+ee[5]) + q1.z*(k1.z+ee[6]) + q1.w*(k1.w+ee[7]);
  #pragma unroll
  for (int off = 1; off < 8; off <<= 1) part += __shfl_xor(part, off, 32);

  float logit = part * 0.125f;   // 1/sqrt(64)
  if ((lane & 7) == 0) {
    int h = lane >> 3;
    logits[(size_t)e * 4 + h] = logit;
    atomic_max_f32(&mbuf[(size_t)dst * 4 + h], logit);
  }
}

// ---------------- edge pass B: p, segment sum, weighted scatter -------------
__global__ __launch_bounds__(256)
void edge_scatter_kernel(const int* __restrict__ eidx, const float* __restrict__ ea,
                         const float* __restrict__ We, const float* __restrict__ be,
                         const float* __restrict__ qkvs, const float* __restrict__ logits,
                         const float* __restrict__ mbuf, float* __restrict__ sbuf,
                         float* __restrict__ accb) {
  const float* sWe = (const float*)dynsmem;
  const float* sbe = sWe + 16 * HCC;
  stage_We_async(We, be);

  int lane = threadIdx.x & 31;
  int e = blockIdx.x * 8 + (threadIdx.x >> 5);
  if (e >= NE) return;
  int src = eidx[e];
  int dst = eidx[NE + e];

  int h = lane >> 3;
  float logit = logits[(size_t)e * 4 + h];
  float mh    = mbuf[(size_t)dst * 4 + h];
  float p     = __expf(logit - mh);
  if ((lane & 7) == 0) atomicAdd(&sbuf[(size_t)dst * 4 + h], p);

  float my = (lane < 16) ? ea[(size_t)e * 16 + lane] : 0.0f;
  float ed[16];
  #pragma unroll
  for (int d = 0; d < 16; ++d) ed[d] = __shfl(my, d, 32);

  int c0 = lane * 8;
  float ee[8];
  edge_embed8(sWe, sbe, ed, c0, ee);

  const float* vrow = qkvs + (size_t)src * QC + 512 + c0;   // V block
  float4 v0 = *(const float4*)vrow, v1 = *(const float4*)(vrow + 4);
  float* ap = accb + (size_t)dst * HCC + c0;
  atomicAdd(ap + 0, p * (v0.x + ee[0]));
  atomicAdd(ap + 1, p * (v0.y + ee[1]));
  atomicAdd(ap + 2, p * (v0.z + ee[2]));
  atomicAdd(ap + 3, p * (v0.w + ee[3]));
  atomicAdd(ap + 4, p * (v1.x + ee[4]));
  atomicAdd(ap + 5, p * (v1.y + ee[5]));
  atomicAdd(ap + 6, p * (v1.z + ee[6]));
  atomicAdd(ap + 7, p * (v1.w + ee[7]));
}

// ---------------- node finalize: normalize + skip + relu --------------------
__global__ void finalize_kernel(const float* __restrict__ accb, const float* __restrict__ sbuf,
                                const float* __restrict__ qkvs, float* __restrict__ hout) {
  int idx = blockIdx.x * blockDim.x + threadIdx.x;
  if (idx >= NN * HCC) return;
  int n = idx >> 8, c = idx & 255, h = c >> 6;
  float s = sbuf[n * 4 + h] + 1e-16f;
  float val = accb[idx] / s + qkvs[(size_t)n * QC + 768 + c];  // skip = x@Ws+bs
  hout[idx] = fmaxf(val, 0.0f);
}

// ---------------- pooling + head MLP ---------------------------------------
__global__ void zero_g_kernel(float* __restrict__ g) {
  int idx = blockIdx.x * blockDim.x + threadIdx.x;
  if (idx < NG * HCC) g[idx] = 0.0f;
}

__global__ void pool_kernel(const float* __restrict__ h, const int* __restrict__ batch,
                            float* __restrict__ g) {
  int idx = blockIdx.x * blockDim.x + threadIdx.x;
  if (idx >= NN * HCC) return;
  int n = idx >> 8, c = idx & 255;
  atomicAdd(&g[(size_t)batch[n] * HCC + c], h[idx]);
}

__global__ __launch_bounds__(128)
void mlp_kernel(const float* __restrict__ g, const float* __restrict__ Wf0,
                const float* __restrict__ bf0, const float* __restrict__ Wf1,
                const float* __restrict__ bf1, float* __restrict__ out) {
  int row = blockIdx.x, j = threadIdx.x;
  const float* gr = g + (size_t)row * HCC;
  float a = bf0[j];
  for (int t = 0; t < HCC; ++t) a += gr[t] * Wf0[t * 128 + j];
  a = fmaxf(a, 0.0f) * Wf1[j];
  __shared__ float red[128];
  red[j] = a; __syncthreads();
  for (int off = 64; off > 0; off >>= 1) { if (j < off) red[j] += red[j + off]; __syncthreads(); }
  if (j == 0) out[row] = red[0] + bf1[0];
}

// ---------------------------------------------------------------------------
extern "C" void kernel_launch(void* const* d_in, const int* in_sizes, int n_in,
                              void* d_out, int out_size, void* d_ws, size_t ws_size,
                              hipStream_t stream) {
  const float* x    = (const float*)d_in[0];
  const int*   eidx = (const int*)d_in[1];
  const float* eattr= (const float*)d_in[2];
  const int*   batch= (const int*)d_in[3];
  // layer 0: Wq0 bq0 Wk0 bk0 Wv0 bv0 Ws0 bs0 We0 be0 -> d_in[4..13]
  // layer 1: -> d_in[14..23]; head: Wf0 bf0 Wf1 bf1 -> d_in[24..27]
  const float* W0[10]; const float* W1[10];
  for (int i = 0; i < 10; ++i) { W0[i] = (const float*)d_in[4 + i]; W1[i] = (const float*)d_in[14 + i]; }
  const float* Wf0 = (const float*)d_in[24];
  const float* bf0 = (const float*)d_in[25];
  const float* Wf1 = (const float*)d_in[26];
  const float* bf1 = (const float*)d_in[27];
  float* out = (float*)d_out;

  // workspace carve (all region sizes are multiples of 32B)
  float* qkvs  = (float*)d_ws;                 // NN*1024
  float* hbuf  = qkvs  + (size_t)NN * QC;      // NN*256
  float* accb  = hbuf  + (size_t)NN * HCC;     // NN*256
  float* mbuf  = accb  + (size_t)NN * HCC;     // NN*4
  float* sbuf  = mbuf  + (size_t)NN * 4;       // NN*4
  float* logit = sbuf  + (size_t)NN * 4;       // NE*4
  _Float16* wpack = (_Float16*)(logit + (size_t)NE * 4);  // 1024*256 halves (max K)
  float* bpack = (float*)(wpack + (size_t)QC * 256);      // 1024
  float* gbuf  = bpack + QC;                               // NG*256

  const int gemm_blocks = (NN + 15) / 16;
  const int node_elems  = NN * HCC;
  const int node_blocks = (node_elems + 255) / 256;
  const int edge_blocks = (NE + 7) / 8;
  const size_t edge_lds = (size_t)(16 * HCC + HCC) * sizeof(float);   // We + be = 17408B

  const float* in_x = x;
  for (int layer = 0; layer < 2; ++layer) {
    const float* const* W = (layer == 0) ? W0 : W1;
    int K = (layer == 0) ? 64 : HCC;
    int kshift = (layer == 0) ? 6 : 8;
    int pack_total = QC * K;
    pack_weights_kernel<<<(pack_total + 255) / 256, 256, 0, stream>>>(
        W[0], W[2], W[4], W[6], W[1], W[3], W[5], W[7], wpack, bpack, K);
    gemm_qkvs_kernel<<<gemm_blocks, 256, (size_t)(16 * K) * sizeof(_Float16), stream>>>(
        in_x, wpack, bpack, qkvs, NN, K, kshift);
    init_layer_kernel<<<node_blocks, 256, 0, stream>>>(mbuf, sbuf, accb);
    edge_logits_kernel<<<edge_blocks, 256, edge_lds, stream>>>(eidx, eattr, W[8], W[9],
                                                               qkvs, logit, mbuf);
    edge_scatter_kernel<<<edge_blocks, 256, edge_lds, stream>>>(eidx, eattr, W[8], W[9],
                                                                qkvs, logit, mbuf, sbuf, accb);
    finalize_kernel<<<node_blocks, 256, 0, stream>>>(accb, sbuf, qkvs, hbuf);
    in_x = hbuf;
  }

  zero_g_kernel<<<(NG * HCC + 255) / 256, 256, 0, stream>>>(gbuf);
  pool_kernel<<<node_blocks, 256, 0, stream>>>(hbuf, batch, gbuf);
  mlp_kernel<<<NG, 128, 0, stream>>>(gbuf, Wf0, bf0, Wf1, bf1, out);
}
